// Cross_Attention_42356967473852
// MI455X (gfx1250) — matrix operand order repaired and checked by reading, not compiled, not run on hardware
//
#include <hip/hip_runtime.h>
#include <hip/hip_bf16.h>

// ---------------------------------------------------------------------------
// Cross-attention with LoRA projections, fused for MI455X (gfx1250, wave32).
//   q  = (z @ wq_down) @ wq_up                [B,Nz,768]   (stored bf16)
//   kv = (x @ wkv_down) @ wkv_up -> K,V       [B,H,Nx,96]  (stored bf16)
//   S  = (q k^T) * (96^-0.5 * temp); P = softmax(S)  (P is output #2, fp32)
//   O  = P V  -> [B,Nz,768] fp32;  out = (O @ wp_down) @ wp_up  (output #1)
// bf16 conversion is pushed into the LoRA producers: the WMMA math is bf16
// regardless, so this halves K/V/Q traffic in the HBM-bound attention kernel
// and removes all cvt work from the QK^T inner loop.
// ---------------------------------------------------------------------------

typedef __attribute__((ext_vector_type(16))) __bf16 v16bf;
typedef __attribute__((ext_vector_type(8)))  __bf16 v8bf;
typedef __attribute__((ext_vector_type(8)))  float  v8f;

#define B_    16
#define NZ_   1024
#define NX_   2048
#define D_    768
#define H_    8
#define HD_   96
#define SROW  (NX_ + 4)      // padded fp32 row stride for S in LDS (2052)

// ---------------------------------------------------------------------------
// Generic dense LoRA:  Y[row,:] = (X[row,:] @ Wd[Din,R]) @ Wu[R,Dout]
// one 256-thread block per row; OutT selects fp32 or bf16 output
// ---------------------------------------------------------------------------
template <int R, typename OutT>
__global__ void lora_dense(const float* __restrict__ X,
                           const float* __restrict__ Wd,
                           const float* __restrict__ Wu,
                           OutT* __restrict__ Y, int Din, int Dout) {
    __shared__ float red[R * 256];
    const int tid = threadIdx.x;
    const long row = blockIdx.x;
    const float* xr = X + row * (long)Din;

    float part[R];
#pragma unroll
    for (int j = 0; j < R; ++j) part[j] = 0.f;
    for (int d = tid; d < Din; d += 256) {
        const float xv = xr[d];
        const float* w = Wd + (long)d * R;
#pragma unroll
        for (int j = 0; j < R; ++j) part[j] += xv * w[j];
    }
#pragma unroll
    for (int j = 0; j < R; ++j) red[j * 256 + tid] = part[j];
    __syncthreads();
    for (int s = 128; s > 0; s >>= 1) {
        if (tid < s) {
#pragma unroll
            for (int j = 0; j < R; ++j) red[j * 256 + tid] += red[j * 256 + tid + s];
        }
        __syncthreads();
    }
    float t[R];
#pragma unroll
    for (int j = 0; j < R; ++j) t[j] = red[j * 256];

    for (int c = tid; c < Dout; c += 256) {
        float acc = 0.f;
#pragma unroll
        for (int j = 0; j < R; ++j) acc += t[j] * Wu[(long)j * Dout + c];
        Y[row * (long)Dout + c] = (OutT)acc;
    }
}

// ---------------------------------------------------------------------------
// kv LoRA (rank 8) with head-major scatter, bf16 output:
//   K[b,h,n,hd], V[b,h,n,hd]   (hd = 96)
// one 256-thread block per x row (row = b*Nx + n)
// ---------------------------------------------------------------------------
__global__ void kv_lora(const float* __restrict__ X,
                        const float* __restrict__ Wd,   // [768,8]
                        const float* __restrict__ Wu,   // [8,1536]
                        __bf16* __restrict__ Kb, __bf16* __restrict__ Vb) {
    __shared__ float red[8 * 256];
    const int tid = threadIdx.x;
    const long row = blockIdx.x;          // b*Nx + n
    const int b = (int)(row / NX_);
    const int n = (int)(row % NX_);
    const float* xr = X + row * (long)D_;

    float part[8];
#pragma unroll
    for (int j = 0; j < 8; ++j) part[j] = 0.f;
    for (int d = tid; d < D_; d += 256) {
        const float xv = xr[d];
        const float* w = Wd + (long)d * 8;
#pragma unroll
        for (int j = 0; j < 8; ++j) part[j] += xv * w[j];
    }
#pragma unroll
    for (int j = 0; j < 8; ++j) red[j * 256 + tid] = part[j];
    __syncthreads();
    for (int s = 128; s > 0; s >>= 1) {
        if (tid < s) {
#pragma unroll
            for (int j = 0; j < 8; ++j) red[j * 256 + tid] += red[j * 256 + tid + s];
        }
        __syncthreads();
    }
    float t[8];
#pragma unroll
    for (int j = 0; j < 8; ++j) t[j] = red[j * 256];

    for (int c = tid; c < 2 * D_; c += 256) {
        float acc = 0.f;
#pragma unroll
        for (int j = 0; j < 8; ++j) acc += t[j] * Wu[(long)j * (2 * D_) + c];
        const int cc = (c < D_) ? c : c - D_;
        const int h = cc / HD_, hd = cc % HD_;
        const long idx = (((long)(b * H_ + h) * NX_) + n) * HD_ + hd;
        if (c < D_) Kb[idx] = (__bf16)acc; else Vb[idx] = (__bf16)acc;
    }
}

// ---------------------------------------------------------------------------
// Fused attention: 1 block = 4 waves = one (b, h, 16-row q tile).
//   LDS: S[16][SROW] fp32 (128 KB) + reduction scratch + O partials.
//   S stage  : 32 col-tiles/wave x 3 wmma (K=96)   -> v_wmma_f32_16x16x32_bf16
//   softmax  : block-wide, 16 rows, attn written once (coalesced)
//   PV stage : 16 K-chunks/wave x 6 N-tiles        -> v_wmma_f32_16x16x32_bf16
// ---------------------------------------------------------------------------
__global__ void __launch_bounds__(128)
attn_fused(const __bf16* __restrict__ Q,     // [B,Nz,768] (head-major columns)
           const __bf16* __restrict__ Kb,    // [B,H,Nx,96]
           const __bf16* __restrict__ Vb,    // [B,H,Nx,96]
           const float* __restrict__ soft_temp,
           float* __restrict__ attn_out,     // [B,H,Nz,Nx]
           float* __restrict__ Oh) {         // [B,Nz,768]
    const int qt  = blockIdx.x;              // 0..63  (16-row q tile)
    const int h   = blockIdx.y;              // 0..7
    const int b   = blockIdx.z;              // 0..15
    const int tid = threadIdx.x;             // 0..127
    const int wave   = tid >> 5;
    const int lane   = tid & 31;
    const int lane16 = lane & 15;
    const int hi     = lane >> 4;            // half-wave select (ISA A/B layouts)

    extern __shared__ float lds[];
    float* S    = lds;                                  // [16][SROW]
    float* red  = lds + 16 * SROW;                      // [128]
    float* ored = red + 128;                            // [4][16][96]

    const float scale = 0.10206207261596575f * soft_temp[0];  // 96^-0.5 * temp
    const int  qrow0  = qt * 16;
    const long qbase  = ((long)(b * NZ_ + qrow0)) * D_ + h * HD_;
    const long kvbase = ((long)(b * H_ + h)) * NX_ * HD_;

    // Warm L2 with this head's V panel (384 KB bf16) while we do S + softmax.
    // Lowered to global_prefetch_b8; no counters, no VGPR results.
    {
        const char* vpan = (const char*)(Vb + kvbase);
        const int panel = NX_ * HD_ * 2;
        for (int off = tid * 128; off < panel; off += 128 * 128)
            __builtin_prefetch(vpan + off, 0, 1);
    }

    // ---- Q A-fragments: 3 chunks of K=32 (16-bit A layout, wave32) -------
    // lane: M = lane%16; VGPR0-7 hold K = hi*16 + 0..15 (this code's contiguous order, not ISA table order).
    v16bf aQ[3];
    {
        const __bf16* qrow = Q + qbase + (long)lane16 * D_;
#pragma unroll
        for (int t = 0; t < 3; ++t) {
            const v8bf lo = *reinterpret_cast<const v8bf*>(qrow + t * 32 + hi * 16);
            const v8bf hv = *reinterpret_cast<const v8bf*>(qrow + t * 32 + hi * 16 + 8);
            aQ[t] = __builtin_shufflevector(lo, hv, 0, 1, 2, 3, 4, 5, 6, 7,
                                            8, 9, 10, 11, 12, 13, 14, 15);
        }
    }

    // ---- S = scale * q k^T : each wave does col-tiles wave, wave+4, ... ---
    for (int ct = wave; ct < NX_ / 16; ct += 4) {
        const int col0 = ct * 16;
        v8f acc = {};
#pragma unroll
        for (int t = 0; t < 3; ++t) {
            // B: lane n = lane%16 holds K = hi*16 + 0..15, same contiguous order as aQ (32B aligned)
            const __bf16* kp = Kb + kvbase + (long)(col0 + lane16) * HD_ + t * 32 + hi * 16;
            const v8bf b0 = *reinterpret_cast<const v8bf*>(kp);
            const v8bf b1 = *reinterpret_cast<const v8bf*>(kp + 8);
            const v16bf bK = __builtin_shufflevector(b0, b1, 0, 1, 2, 3, 4, 5, 6, 7,
                                                     8, 9, 10, 11, 12, 13, 14, 15);
            acc = __builtin_amdgcn_wmma_f32_16x16x32_bf16(
                false, aQ[t], false, bK, (short)0, acc, false, false);
        }
        // C layout: VGPR r -> M = r + hi*8, N = lane%16
#pragma unroll
        for (int r = 0; r < 8; ++r)
            S[(r + hi * 8) * SROW + col0 + lane16] = acc[r] * scale;
    }
    __syncthreads();

    // ---- softmax per row, write normalized attn once ----------------------
    for (int m = 0; m < 16; ++m) {
        float* srow = S + m * SROW;
        float mx = -3.402823466e38f;
        for (int c = tid; c < NX_; c += 128) mx = fmaxf(mx, srow[c]);
        red[tid] = mx; __syncthreads();
        for (int s = 64; s > 0; s >>= 1) {
            if (tid < s) red[tid] = fmaxf(red[tid], red[tid + s]);
            __syncthreads();
        }
        const float rowmax = red[0]; __syncthreads();

        float sm = 0.f;
        for (int c = tid; c < NX_; c += 128) {
            const float e = __expf(srow[c] - rowmax);
            srow[c] = e; sm += e;
        }
        red[tid] = sm; __syncthreads();
        for (int s = 64; s > 0; s >>= 1) {
            if (tid < s) red[tid] += red[tid + s];
            __syncthreads();
        }
        const float inv = 1.f / red[0]; __syncthreads();

        float* ap = attn_out + (((long)(b * H_ + h) * NZ_) + qrow0 + m) * (long)NX_;
        for (int c = tid; c < NX_; c += 128) {
            const float p = srow[c] * inv;
            srow[c] = p;
            ap[c] = p;
        }
        __syncthreads();
    }

    // ---- O = P V : K-dim split over 4 waves, 6 N-tiles of 16 --------------
    v8f oc[6];
#pragma unroll
    for (int nt = 0; nt < 6; ++nt) oc[nt] = (v8f){};
    for (int ch = wave; ch < NX_ / 32; ch += 4) {
        const int c0 = ch * 32;
        // A fragment (P) gathered from LDS, element e = k hi*16 + e (contiguous order) (fp32 -> bf16)
        v16bf aP;
        {
            const float* pp = S + lane16 * SROW + c0 + hi * 16;
#pragma unroll
            for (int e = 0; e < 8; ++e) {
                aP[e]     = (__bf16)pp[e];
                aP[8 + e] = (__bf16)pp[8 + e];
            }
        }
#pragma unroll
        for (int nt = 0; nt < 6; ++nt) {
            // B: lane half hi holds rows c0+hi*16+e, same k order as aP (cols contiguous across lanes)
            const __bf16* vp = Vb + kvbase + (long)(c0 + hi * 16) * HD_ + nt * 16 + lane16;
            v16bf bV;
#pragma unroll
            for (int e = 0; e < 16; ++e) bV[e] = vp[(long)e * HD_];
            oc[nt] = __builtin_amdgcn_wmma_f32_16x16x32_bf16(
                false, aP, false, bV, (short)0, oc[nt], false, false);
        }
    }

    // ---- cross-wave partial reduction, write O heads ----------------------
#pragma unroll
    for (int nt = 0; nt < 6; ++nt)
#pragma unroll
        for (int r = 0; r < 8; ++r)
            ored[(wave * 16 + (r + hi * 8)) * HD_ + nt * 16 + lane16] = oc[nt][r];
    __syncthreads();
    for (int i = tid; i < 16 * HD_; i += 128) {
        const float s = ored[i] + ored[16 * HD_ + i] + ored[2 * 16 * HD_ + i] + ored[3 * 16 * HD_ + i];
        const int m = i / HD_, c = i % HD_;
        Oh[((long)(b * NZ_ + qrow0 + m)) * D_ + h * HD_ + c] = s;
    }
}

// ---------------------------------------------------------------------------
extern "C" void kernel_launch(void* const* d_in, const int* in_sizes, int n_in,
                              void* d_out, int out_size, void* d_ws, size_t ws_size,
                              hipStream_t stream) {
    const float* z        = (const float*)d_in[0];
    const float* x        = (const float*)d_in[1];
    const float* wq_down  = (const float*)d_in[2];
    const float* wq_up    = (const float*)d_in[3];
    const float* wkv_down = (const float*)d_in[4];
    const float* wkv_up   = (const float*)d_in[5];
    const float* wp_down  = (const float*)d_in[6];
    const float* wp_up    = (const float*)d_in[7];
    const float* soft_t   = (const float*)d_in[8];

    float* out  = (float*)d_out;                                 // [B,Nz,768]
    float* attn = out + (size_t)B_ * NZ_ * D_;                   // [B,H,Nz,Nx]

    char* ws = (char*)d_ws;
    const size_t qBytes  = (size_t)B_ * NZ_ * D_ * sizeof(__bf16);        // 25.2 MB
    const size_t kvBytes = (size_t)B_ * H_ * NX_ * HD_ * sizeof(__bf16);  // 50.3 MB
    __bf16* q  = (__bf16*)ws;
    __bf16* Kb = (__bf16*)(ws + qBytes);
    __bf16* Vb = (__bf16*)(ws + qBytes + kvBytes);
    float*  Oh = (float*)(ws + qBytes + 2 * kvBytes);                     // 50.3 MB fp32

    // 1) LoRA projections (bf16 producers for the WMMA consumers)
    lora_dense<4, __bf16><<<B_ * NZ_, 256, 0, stream>>>(z, wq_down, wq_up, q, D_, D_);
    kv_lora<<<B_ * NX_, 256, 0, stream>>>(x, wkv_down, wkv_up, Kb, Vb);

    // 2) fused attention (dyn LDS ~153 KB, fine on 320 KB/WGP CDNA5)
    const size_t shmem = (size_t)(16 * SROW + 128 + 4 * 16 * HD_) * sizeof(float);
    hipFuncSetAttribute((const void*)attn_fused,
                        hipFuncAttributeMaxDynamicSharedMemorySize, (int)shmem);
    attn_fused<<<dim3(NZ_ / 16, H_, B_), 128, shmem, stream>>>(
        q, Kb, Vb, soft_t, attn, Oh);

    // 3) output LoRA (fp32 output)
    lora_dense<4, float><<<B_ * NZ_, 256, 0, stream>>>(Oh, wp_down, wp_up, out, D_, D_);
}
